// Model_39676907885299
// MI455X (gfx1250) — matrix operand order repaired and checked by reading, not compiled, not run on hardware
//
#include <hip/hip_runtime.h>
#include <hip/hip_bf16.h>

typedef _Float16 half_t;
typedef __attribute__((ext_vector_type(16))) _Float16 v16h;
typedef __attribute__((ext_vector_type(8)))  _Float16 v8h;
typedef __attribute__((ext_vector_type(8)))  float    v8f;

#define B_  4
#define L_  2048
#define D_  512
#define H_  8
#define DK_ 64
#define DV_ 64
#define MROWS_ (B_ * L_)          // 8192
#define NDIM_  (H_ * DK_)         // 512

__device__ __forceinline__ v8f wmma16x16x32(v16h a, v16h b, v8f c) {
  // D = A(16x32,f16) * B(32x16,f16) + C(16x16,f32)
  return __builtin_amdgcn_wmma_f32_16x16x32_f16(false, a, false, b, (short)0, c,
                                                false, false);
}

__device__ __forceinline__ v16h cat8(v8h lo, v8h hi) {
  return __builtin_shufflevector(lo, hi, 0, 1, 2, 3, 4, 5, 6, 7, 8, 9, 10, 11,
                                 12, 13, 14, 15);
}

// A-fragment (16x32 f16): lane half h holds K = h*8+0..7 then 16+h*8+0..7,
// both contiguous 16B runs from the lane's row (ISA 7.12.2).
__device__ __forceinline__ v16h load_afrag(const half_t* rowPtr, int lh) {
  v8h lo = *(const v8h*)(rowPtr + lh * 8);
  v8h hi = *(const v8h*)(rowPtr + 16 + lh * 8);
  return cat8(lo, hi);
}

// B-fragment (32x16 f16) from a K-major (transposed) LDS row: lane half h
// holds K = h*8+0..7 then 16+h*8+0..7 (matches A-fragment) = two b128 loads.
__device__ __forceinline__ v16h load_bfrag_t(const half_t* rowPtr, int lh) {
  v8h lo = *(const v8h*)(rowPtr + lh * 8);
  v8h hi = *(const v8h*)(rowPtr + 16 + lh * 8);
  return cat8(lo, hi);
}

// Async global->LDS copy (16B per lane), tracked by ASYNCcnt.
// ldsAddr: wave-relative LDS byte address (low 32 bits of flat shared ptr).
__device__ __forceinline__ void async_copy_b128(unsigned int ldsAddr,
                                                const void* gptr) {
  asm volatile("global_load_async_to_lds_b128 %0, %1, off"
               :
               : "v"(ldsAddr), "v"(gptr)
               : "memory");
}

__device__ __forceinline__ void wait_asynccnt0() {
  asm volatile("s_wait_asynccnt 0x0" ::: "memory");
}

__device__ __forceinline__ unsigned int lds_addr_of(const void* p) {
  return (unsigned int)(unsigned long long)p;  // flat shared addr low bits
}

// ---------------------------------------------------------------------------
// Weight prep: W f32 [K][N] -> WT f16 [N][K] (transposed, converted) so GEMM
// B-tiles stage with fully vectorized global + LDS traffic.
// ---------------------------------------------------------------------------
__global__ void __launch_bounds__(256)
transpose_convert_kernel(const float* __restrict__ W, half_t* __restrict__ WT,
                         int K, int N) {
  constexpr int LDT = 68;  // f32 per row, 272B (16B multiple)
  __shared__ __align__(16) float T[64 * LDT];
  const int tid = threadIdx.x;
  const int kbase = blockIdx.y * 64;
  const int nbase = blockIdx.x * 64;
#pragma unroll
  for (int i = 0; i < 4; ++i) {
    int chunk = tid + i * 256;   // 1024 chunks of float4
    int row = chunk >> 4;
    int c4 = (chunk & 15) * 4;
    *(float4*)&T[row * LDT + c4] =
        *(const float4*)(W + (size_t)(kbase + row) * N + nbase + c4);
  }
  __syncthreads();
#pragma unroll
  for (int i = 0; i < 2; ++i) {
    int chunk = tid + i * 256;   // 512 chunks of 8 halfs
    int n = chunk >> 3;
    int kc = (chunk & 7) * 8;
    v8h st;
#pragma unroll
    for (int j = 0; j < 8; ++j) st[j] = (half_t)T[(kc + j) * LDT + n];
    *(v8h*)(WT + (size_t)(nbase + n) * K + kbase + kc) = st;
  }
}

// ---------------------------------------------------------------------------
// Tiled WMMA GEMM: out[M,N] = A[M,K] * W[K,N] + bias[N], weights given as
// WT f16 [N][K]. block = 128 threads (4 waves), tile 64x64, k-step 32.
// Register double-buffering: global loads for step k+1 in flight during the
// WMMAs of step k.
// OUT_MODE: 0 = f32 row-major, 1 = f16 row-major,
//           2 = f16 "V-transposed" [B][H][DV][L] (for attention V feed).
// ---------------------------------------------------------------------------
template <bool A_HALF, int OUT_MODE>
__global__ void __launch_bounds__(128)
gemm_wmma_kernel(const void* __restrict__ Av, const half_t* __restrict__ WT,
                 const float* __restrict__ bias, void* __restrict__ Ov,
                 int M, int N, int K) {
  constexpr int LDA = 40;  // halfs per row (80B, 16B multiple)
  __shared__ __align__(16) half_t As[64 * LDA];
  __shared__ __align__(16) half_t BsT[64 * LDA];  // n-major: [n][kk]

  const int tid  = threadIdx.x;
  const int wave = tid >> 5;
  const int lane = tid & 31;
  const int ln   = lane & 15;
  const int lh   = lane >> 4;
  const int mbase = blockIdx.y * 64;
  const int nbase = blockIdx.x * 64;

  // chunk geometry: 2 chunks of 8 halfs per thread for each of A/B tiles
  int rowC[2], colC[2];
#pragma unroll
  for (int i = 0; i < 2; ++i) {
    int chunk = tid + i * 128;
    rowC[i] = chunk >> 2;
    colC[i] = (chunk & 3) * 8;
  }

  v8h aReg[2], bReg[2];
  auto load_step = [&](int k0) {
#pragma unroll
    for (int i = 0; i < 2; ++i) {
      if (A_HALF) {
        aReg[i] = *(const v8h*)((const half_t*)Av +
                                (size_t)(mbase + rowC[i]) * K + k0 + colC[i]);
      } else {
        const float* src =
            (const float*)Av + (size_t)(mbase + rowC[i]) * K + k0 + colC[i];
        float4 f0 = *(const float4*)(src);
        float4 f1 = *(const float4*)(src + 4);
        v8h st;
        st[0] = (half_t)f0.x; st[1] = (half_t)f0.y;
        st[2] = (half_t)f0.z; st[3] = (half_t)f0.w;
        st[4] = (half_t)f1.x; st[5] = (half_t)f1.y;
        st[6] = (half_t)f1.z; st[7] = (half_t)f1.w;
        aReg[i] = st;
      }
      bReg[i] = *(const v8h*)(WT + (size_t)(nbase + rowC[i]) * K + k0 + colC[i]);
    }
  };

  v8f acc[4] = {};
  load_step(0);

  for (int k0 = 0; k0 < K; k0 += 32) {
    __syncthreads();  // prior compute done reading LDS
#pragma unroll
    for (int i = 0; i < 2; ++i) {
      *(v8h*)&As[rowC[i] * LDA + colC[i]] = aReg[i];
      *(v8h*)&BsT[rowC[i] * LDA + colC[i]] = bReg[i];
    }
    __syncthreads();
    if (k0 + 32 < K) load_step(k0 + 32);  // in flight during WMMAs below

    v16h a = load_afrag(&As[(wave * 16 + ln) * LDA], lh);
#pragma unroll
    for (int ns = 0; ns < 4; ++ns) {
      v16h b = load_bfrag_t(&BsT[(ns * 16 + ln) * LDA], lh);
      acc[ns] = wmma16x16x32(a, b, acc[ns]);
    }
  }

  // Epilogue: C/D layout — lane (n=ln, h=lh), VGPR r -> row m = r + 8h
#pragma unroll
  for (int ns = 0; ns < 4; ++ns) {
    int col = nbase + ns * 16 + ln;
    float bv = bias[col];
#pragma unroll
    for (int r = 0; r < 8; ++r) {
      int row = mbase + wave * 16 + r + 8 * lh;
      float val = acc[ns][r] + bv;
      if (OUT_MODE == 0) {
        ((float*)Ov)[(size_t)row * N + col] = val;
      } else if (OUT_MODE == 1) {
        ((half_t*)Ov)[(size_t)row * N + col] = (half_t)val;
      } else {
        // V fed to attention as [B][H][DV][L]
        int b = row >> 11;          // / L_
        int l = row & (L_ - 1);
        int h = col >> 6;           // / DV_
        int dv = col & (DV_ - 1);
        ((half_t*)Ov)[(((size_t)b * H_ + h) * DV_ + dv) * L_ + l] = (half_t)val;
      }
    }
  }
}

// ---------------------------------------------------------------------------
// Flash attention: block = one 64-row q-tile of one (head, batch), 4 waves.
// K/V tiles double-buffered in LDS via GLOBAL_LOAD_ASYNC_TO_LDS_B128
// (ASYNCcnt); one s_wait_asynccnt + one barrier per key tile.
// Register-resident online softmax via cross-lane shuffles.
// ---------------------------------------------------------------------------
__global__ void __launch_bounds__(128)
flash_attn_kernel(const half_t* __restrict__ Qh,   // [B, L, H*DK]
                  const half_t* __restrict__ Kh,   // [B, L, H*DK]
                  const half_t* __restrict__ Vt,   // [B, H, DV, L]
                  const float* __restrict__ mask,  // [L, L]
                  half_t* __restrict__ Xo) {       // scrambled [H,Lq,B,Dv] flat
  constexpr int LDQ = 72;  // halfs per row (144B, 16B multiple)
  constexpr int NT = L_ / 64;
  __shared__ __align__(16) half_t Qs[64 * LDQ];
  __shared__ __align__(16) half_t Ks[2][64 * LDQ];  // [key][dk], double-buffered
  __shared__ __align__(16) half_t Vs[2][64 * LDQ];  // [dv][key], double-buffered
  __shared__ __align__(16) half_t Ps[64 * LDQ];     // [q][key]

  const int tid  = threadIdx.x;
  const int wave = tid >> 5;
  const int lane = tid & 31;
  const int ln   = lane & 15;
  const int lh   = lane >> 4;
  const int qbase = blockIdx.x * 64;
  const int hh = blockIdx.y % H_;
  const int b  = blockIdx.y / H_;

  const half_t* Khead  = Kh + (size_t)b * L_ * NDIM_ + hh * DK_;
  const half_t* Vthead = Vt + ((size_t)b * H_ + hh) * DV_ * L_;

  // chunk geometry for 64x64 f16 tiles: 4 chunks of 8 halfs per thread
  int rowC[4], colC[4];
  unsigned int ldsOffT[4];  // byte offset of chunk within a tile buffer
#pragma unroll
  for (int i = 0; i < 4; ++i) {
    int chunk = tid + i * 128;
    rowC[i] = chunk >> 3;
    colC[i] = (chunk & 7) * 8;
    ldsOffT[i] = (unsigned int)((rowC[i] * LDQ + colC[i]) * sizeof(half_t));
  }
  const unsigned int ldsK[2] = {lds_addr_of(&Ks[0][0]), lds_addr_of(&Ks[1][0])};
  const unsigned int ldsV[2] = {lds_addr_of(&Vs[0][0]), lds_addr_of(&Vs[1][0])};

  auto issue_tile = [&](int kt, int buf) {
#pragma unroll
    for (int i = 0; i < 4; ++i) {
      async_copy_b128(ldsK[buf] + ldsOffT[i],
                      Khead + (size_t)(kt * 64 + rowC[i]) * NDIM_ + colC[i]);
      async_copy_b128(ldsV[buf] + ldsOffT[i],
                      Vthead + (size_t)rowC[i] * L_ + kt * 64 + colC[i]);
    }
  };

  // Stage Q tile (64 x 64 f16) synchronously; kick off first K/V tile async.
  issue_tile(0, 0);
#pragma unroll
  for (int i = 0; i < 4; ++i) {
    *(v8h*)&Qs[rowC[i] * LDQ + colC[i]] = *(const v8h*)(
        Qh + (size_t)(b * L_ + qbase + rowC[i]) * NDIM_ + hh * DK_ + colC[i]);
  }
  __syncthreads();

  v16h aq[2];
#pragma unroll
  for (int c = 0; c < 2; ++c)
    aq[c] = load_afrag(&Qs[(wave * 16 + ln) * LDQ + c * 32], lh);

  v8f o[4] = {};
  float rowM[8], rowL[8];  // per-lane running stats for rows r + 8*lh
#pragma unroll
  for (int r = 0; r < 8; ++r) { rowM[r] = -3.0e38f; rowL[r] = 0.0f; }

  int cur = 0;
  for (int kt = 0; kt < NT; ++kt) {
    wait_asynccnt0();   // this wave's async copies for tile kt landed
    __syncthreads();    // all waves' copies landed; prior compute done
    if (kt + 1 < NT) issue_tile(kt + 1, cur ^ 1);  // overlap with compute

    const half_t* KsC = &Ks[cur][0];
    const half_t* VsC = &Vs[cur][0];

    // --- S = Q K^T : K tile rows are keys, contiguous along dk ---
    v8f s[4];
#pragma unroll
    for (int ns = 0; ns < 4; ++ns) {
      const half_t* kp = &KsC[(ns * 16 + ln) * LDQ];
      v8f t = {};
      t = wmma16x16x32(aq[0], load_bfrag_t(kp, lh), t);
      t = wmma16x16x32(aq[1], load_bfrag_t(kp + 32, lh), t);
      s[ns] = t;
    }

    // scale + additive mask
    const float scl = 0.125f;  // 1/sqrt(64)
#pragma unroll
    for (int ns = 0; ns < 4; ++ns) {
#pragma unroll
      for (int r = 0; r < 8; ++r) {
        int qrow = qbase + wave * 16 + r + 8 * lh;
        int kcol = kt * 64 + ns * 16 + ln;
        s[ns][r] = s[ns][r] * scl + mask[(size_t)qrow * L_ + kcol];
      }
    }

    // --- register online softmax: row spans 16 lanes of the half-group ---
#pragma unroll
    for (int r = 0; r < 8; ++r) {
      float mt = fmaxf(fmaxf(s[0][r], s[1][r]), fmaxf(s[2][r], s[3][r]));
      mt = fmaxf(mt, __shfl_xor(mt, 1, 32));
      mt = fmaxf(mt, __shfl_xor(mt, 2, 32));
      mt = fmaxf(mt, __shfl_xor(mt, 4, 32));
      mt = fmaxf(mt, __shfl_xor(mt, 8, 32));
      float mnew = fmaxf(rowM[r], mt);
      float corr = __expf(rowM[r] - mnew);
      rowM[r] = mnew;
      float sum = 0.0f;
#pragma unroll
      for (int ns = 0; ns < 4; ++ns) {
        float p = __expf(s[ns][r] - mnew);
        s[ns][r] = p;
        sum += p;
      }
      sum += __shfl_xor(sum, 1, 32);
      sum += __shfl_xor(sum, 2, 32);
      sum += __shfl_xor(sum, 4, 32);
      sum += __shfl_xor(sum, 8, 32);
      rowL[r] = rowL[r] * corr + sum;
#pragma unroll
      for (int ns = 0; ns < 4; ++ns) o[ns][r] *= corr;
    }

    // spill P (f16) to wave-private LDS rows for the C->A layout remap
#pragma unroll
    for (int ns = 0; ns < 4; ++ns) {
#pragma unroll
      for (int r = 0; r < 8; ++r) {
        Ps[(wave * 16 + r + 8 * lh) * LDQ + ns * 16 + ln] = (half_t)s[ns][r];
      }
    }

    // --- O += P V : A = P, B tile rows are dv, contiguous along key ---
    v16h ap[2];
#pragma unroll
    for (int c = 0; c < 2; ++c)
      ap[c] = load_afrag(&Ps[(wave * 16 + ln) * LDQ + c * 32], lh);
#pragma unroll
    for (int ns = 0; ns < 4; ++ns) {
      const half_t* vp = &VsC[(ns * 16 + ln) * LDQ];
      o[ns] = wmma16x16x32(ap[0], load_bfrag_t(vp, lh), o[ns]);
      o[ns] = wmma16x16x32(ap[1], load_bfrag_t(vp + 32, lh), o[ns]);
    }
    cur ^= 1;
  }

  // Epilogue: normalize (stats live in registers) and store scrambled layout
#pragma unroll
  for (int r = 0; r < 8; ++r) {
    float rl = 1.0f / rowL[r];
    int q = qbase + wave * 16 + r + 8 * lh;
#pragma unroll
    for (int ns = 0; ns < 4; ++ns) {
      int d = ns * 16 + ln;
      size_t idx = (size_t)hh * (L_ * B_ * DV_) + (size_t)q * (B_ * DV_) +
                   (size_t)b * DV_ + d;
      Xo[idx] = (half_t)(o[ns][r] * rl);
    }
  }
}

// ---------------------------------------------------------------------------
extern "C" void kernel_launch(void* const* d_in, const int* in_sizes, int n_in,
                              void* d_out, int out_size, void* d_ws,
                              size_t ws_size, hipStream_t stream) {
  const float* query = (const float*)d_in[0];
  const float* key   = (const float*)d_in[1];
  const float* value = (const float*)d_in[2];
  const float* mask  = (const float*)d_in[3];
  const float* Wq = (const float*)d_in[4];
  const float* bq = (const float*)d_in[5];
  const float* Wk = (const float*)d_in[6];
  const float* bk = (const float*)d_in[7];
  const float* Wv = (const float*)d_in[8];
  const float* bv = (const float*)d_in[9];
  const float* Wo = (const float*)d_in[10];
  const float* bo = (const float*)d_in[11];

  const size_t act = (size_t)MROWS_ * NDIM_;  // 8192*512 elements
  const size_t wel = (size_t)D_ * NDIM_;      // 512*512 elements
  half_t* Qh  = (half_t*)d_ws;
  half_t* Kh  = Qh + act;
  half_t* Vth = Kh + act;   // [B,H,DV,L]
  half_t* Xo  = Vth + act;
  half_t* WqT = Xo + act;
  half_t* WkT = WqT + wel;
  half_t* WvT = WkT + wel;
  half_t* WoT = WvT + wel;  // total ~34 MB workspace

  dim3 blkT(256), grdT(NDIM_ / 64, D_ / 64);  // (8, 8)
  transpose_convert_kernel<<<grdT, blkT, 0, stream>>>(Wq, WqT, D_, NDIM_);
  transpose_convert_kernel<<<grdT, blkT, 0, stream>>>(Wk, WkT, D_, NDIM_);
  transpose_convert_kernel<<<grdT, blkT, 0, stream>>>(Wv, WvT, D_, NDIM_);
  transpose_convert_kernel<<<grdT, blkT, 0, stream>>>(Wo, WoT, NDIM_, D_);

  dim3 blk(128);
  dim3 grdG(NDIM_ / 64, MROWS_ / 64);  // (8, 128)
  gemm_wmma_kernel<false, 1><<<grdG, blk, 0, stream>>>(
      (const void*)query, WqT, bq, (void*)Qh, MROWS_, NDIM_, D_);
  gemm_wmma_kernel<false, 1><<<grdG, blk, 0, stream>>>(
      (const void*)key, WkT, bk, (void*)Kh, MROWS_, NDIM_, D_);
  gemm_wmma_kernel<false, 2><<<grdG, blk, 0, stream>>>(
      (const void*)value, WvT, bv, (void*)Vth, MROWS_, NDIM_, D_);

  dim3 grdA(L_ / 64, B_ * H_);  // (32, 32)
  flash_attn_kernel<<<grdA, blk, 0, stream>>>(Qh, Kh, Vth, mask, Xo);

  gemm_wmma_kernel<true, 0><<<grdG, blk, 0, stream>>>(
      (const void*)Xo, WoT, bo, d_out, MROWS_, D_, NDIM_);
}
